// WindowAttention_43233140801761
// MI455X (gfx1250) — compile-verified, matrix operand order checked by reading, not run on hardware
//
#include <hip/hip_runtime.h>
#include <hip/hip_bf16.h>

typedef __attribute__((ext_vector_type(16))) _Float16 v16h;
typedef __attribute__((ext_vector_type(8)))  _Float16 v8h;
typedef __attribute__((ext_vector_type(4)))  _Float16 v4h;
typedef __attribute__((ext_vector_type(8)))  float    v8f;
typedef __attribute__((ext_vector_type(4)))  float    v4f;

#define DIM   192
#define NH    6
#define HD    32
#define NTOK  49
#define NPAD  64
#define BWIN  4096
#define NW    64

// workspace layout (bytes), all 256-aligned
constexpr size_t OFF_WQ   = 0;                          // 576*192 f16   = 221184
constexpr size_t OFF_WP   = 221184;                     // 192*192 f16   = 73728
constexpr size_t OFF_COMB = 294912;                     // 64*6*64*64 f32= 6291456
constexpr size_t OFF_Q    = 6586368;                    // 4096*6*64*32 f16
constexpr size_t OFF_K    = OFF_Q  + 100663296;
constexpr size_t OFF_VT   = OFF_K  + 100663296;
constexpr size_t OFF_AOUT = OFF_VT + 100663296;         // 200704*192 f16

__device__ __forceinline__ v16h load_frag(const _Float16* p) {
    v8h a0 = *(const v8h*)p;
    v8h a1 = *(const v8h*)(p + 16);
    return __builtin_shufflevector(a0, a1, 0,1,2,3,4,5,6,7,8,9,10,11,12,13,14,15);
}

__device__ __forceinline__ v8f wmma_f16(v16h a, v16h b, v8f c) {
    return __builtin_amdgcn_wmma_f32_16x16x32_f16(false, a, false, b, (short)0, c, false, false);
}

// ---------------- kernel 0: weight convert + combined bias table ----------------
__global__ __launch_bounds__(256) void setup_kernel(
    const float* __restrict__ qkv_w, const float* __restrict__ proj_w,
    const float* __restrict__ rpb, const int* __restrict__ rel_idx,
    const float* __restrict__ mask,
    _Float16* __restrict__ wq16, _Float16* __restrict__ wp16, float* __restrict__ comb)
{
    int idx = blockIdx.x * 256 + threadIdx.x;
    if (idx < 3 * DIM * DIM) wq16[idx] = (_Float16)qkv_w[idx];
    if (idx < DIM * DIM)     wp16[idx] = (_Float16)proj_w[idx];
    if (idx < NW * NH * NPAD * NPAD) {
        int m  = idx & 63;
        int n  = (idx >> 6) & 63;
        int hw = idx >> 12;           // w*6 + h
        int h  = hw % NH;
        int w  = hw / NH;
        float v;
        if (n >= NTOK)       v = -1e30f;              // padded key columns -> softmax 0
        else if (m >= NTOK)  v = 0.0f;                // padded query rows (discarded later)
        else v = rpb[rel_idx[m * NTOK + n] * NH + h] + mask[((size_t)w * NTOK + m) * NTOK + n];
        comb[idx] = v;
    }
}

// ---------------- kernel 1: per-window QKV projection -> Q,K,VT f16 ----------------
__global__ __launch_bounds__(192) void qkv_kernel(
    const float* __restrict__ x, const _Float16* __restrict__ w16,
    const float* __restrict__ qkv_b,
    _Float16* __restrict__ Qb, _Float16* __restrict__ Kb, _Float16* __restrict__ VTb)
{
    __shared__ _Float16 xs[NPAD * DIM];            // 24.6 KB
    const int b    = blockIdx.x;
    const int tid  = threadIdx.x;                  // 0..191
    // stage x window as f16, zero-pad rows >= 49
    for (int row = 0; row < NPAD; ++row) {
        float v = (row < NTOK) ? x[((size_t)b * NTOK + row) * DIM + tid] : 0.0f;
        xs[row * DIM + tid] = (_Float16)v;
    }
    __syncthreads();

    const int wave = tid >> 5, lane = tid & 31;
    const int lh = lane & 15, hi = lane >> 4;
    const float scale = 0.17677669529663687f;      // 32^-0.5

    for (int ntl = 0; ntl < 6; ++ntl) {
        const int nt = wave * 6 + ntl;             // 0..35
        const int o  = nt * 16 + lh;               // output feature (B column)
        v8f acc[4] = {v8f{}, v8f{}, v8f{}, v8f{}};
        #pragma unroll
        for (int kt = 0; kt < 6; ++kt) {
            const int k0 = kt * 32 + hi * 8;
            v16h Bf = load_frag(w16 + (size_t)o * DIM + k0);
            #pragma unroll
            for (int mt = 0; mt < 4; ++mt) {
                v16h Af = load_frag(xs + (mt * 16 + lh) * DIM + k0);
                acc[mt] = wmma_f16(Af, Bf, acc[mt]);
            }
        }
        const float bias = qkv_b[o];
        const int c = o / DIM, od = o % DIM, h = od / HD, d = od % HD;
        const size_t hb = (size_t)b * NH + h;
        if (c == 2) {                              // V: transposed store [b][h][d][tok], natural frag order
            #pragma unroll
            for (int mt = 0; mt < 4; ++mt) {
                const int m0 = mt * 16 + hi * 8;
                v8h pk;
                #pragma unroll
                for (int j = 0; j < 8; ++j) pk[j] = (_Float16)(acc[mt][j] + bias);
                *(v8h*)(VTb + (hb * HD + d) * NPAD + m0) = pk;
            }
        } else {                                   // Q/K: [b][h][tok][d]
            _Float16* dst = (c == 0 ? Qb : Kb) + hb * NPAD * HD + d;
            const float s = (c == 0) ? scale : 1.0f;
            #pragma unroll
            for (int mt = 0; mt < 4; ++mt) {
                const int m0 = mt * 16 + hi * 8;
                #pragma unroll
                for (int j = 0; j < 8; ++j)
                    dst[(size_t)(m0 + j) * HD] = (_Float16)((acc[mt][j] + bias) * s);
            }
        }
    }
}

// ---------------- kernel 2: attention core, one wave per (window, head) ----------------
__global__ __launch_bounds__(64) void attn_kernel(
    const _Float16* __restrict__ Qb, const _Float16* __restrict__ Kb,
    const _Float16* __restrict__ VTb, const float* __restrict__ comb,
    _Float16* __restrict__ aout)
{
    __shared__ float    sS[2][NPAD * NPAD];        // 2 x 16 KB scores
    __shared__ _Float16 sP[2][NPAD * NPAD];        // 2 x 8 KB probs (f16)
    __shared__ float    sR[2][NPAD];               // 1/rowsum
    const int wave = threadIdx.x >> 5, lane = threadIdx.x & 31;
    const int lh = lane & 15, hi = lane >> 4;
    const int gid = blockIdx.x * 2 + wave;         // b*6 + h
    const int b = gid / NH, h = gid % NH, w = b & (NW - 1);
    float*    S = sS[wave];
    _Float16* P = sP[wave];
    float*    R = sR[wave];
    const size_t hb = (size_t)gid;
    const int k0 = hi * 8;

    // Q fragments (resident)
    v16h Qf[4];
    const _Float16* qbase = Qb + hb * NPAD * HD;
    #pragma unroll
    for (int mt = 0; mt < 4; ++mt)
        Qf[mt] = load_frag(qbase + (mt * 16 + lh) * HD + k0);

    // scores = Q K^T + comb, streamed to LDS
    const float* combh = comb + ((size_t)(w * NH + h)) * NPAD * NPAD;
    const _Float16* kbase = Kb + hb * NPAD * HD;
    #pragma unroll
    for (int nt = 0; nt < 4; ++nt) {
        v16h Kf = load_frag(kbase + (nt * 16 + lh) * HD + k0);
        const int n = nt * 16 + lh;
        #pragma unroll
        for (int mt = 0; mt < 4; ++mt) {
            v8f acc = {};
            acc = wmma_f16(Qf[mt], Kf, acc);
            const int m0 = mt * 16 + hi * 8;
            const float* cp = combh + (size_t)n * NPAD + m0;
            v4f c0 = *(const v4f*)cp, c1 = *(const v4f*)(cp + 4);
            #pragma unroll
            for (int j = 0; j < 8; ++j)
                S[(m0 + j) * NPAD + n] = acc[j] + (j < 4 ? c0[j] : c1[j - 4]);
        }
    }
    __syncthreads();

    // rowwise softmax: each lane owns rows lane and lane+32
    #pragma unroll
    for (int r = 0; r < 2; ++r) {
        const int m = lane + r * 32;
        const float* row = S + m * NPAD;
        float mx = -1e30f;
        #pragma unroll
        for (int i = 0; i < 16; ++i) {
            v4f t = *(const v4f*)(row + i * 4);
            mx = fmaxf(mx, fmaxf(fmaxf(t[0], t[1]), fmaxf(t[2], t[3])));
        }
        float sum = 0.0f;
        _Float16* prow = P + m * NPAD;
        #pragma unroll
        for (int i = 0; i < 16; ++i) {
            v4f t = *(const v4f*)(row + i * 4);
            float e0 = __expf(t[0] - mx), e1 = __expf(t[1] - mx);
            float e2 = __expf(t[2] - mx), e3 = __expf(t[3] - mx);
            sum += (e0 + e1) + (e2 + e3);
            v4h pk; pk[0] = (_Float16)e0; pk[1] = (_Float16)e1;
            pk[2] = (_Float16)e2; pk[3] = (_Float16)e3;
            *(v4h*)(prow + i * 4) = pk;
        }
        R[m] = 1.0f / sum;
    }
    __syncthreads();

    // out = P @ V  (B operand = VT rows), normalize, store f16 token rows
    const _Float16* vtb = VTb + hb * HD * NPAD;
    #pragma unroll
    for (int dt = 0; dt < 2; ++dt) {
        const int d = dt * 16 + lh;
        v8f acc[4] = {v8f{}, v8f{}, v8f{}, v8f{}};
        #pragma unroll
        for (int kt = 0; kt < 2; ++kt) {
            const int kk = kt * 32 + k0;
            v16h Bf = load_frag(vtb + (size_t)d * NPAD + kk);
            #pragma unroll
            for (int mt = 0; mt < 4; ++mt) {
                v16h Af = load_frag(P + (mt * 16 + lh) * NPAD + kk);
                acc[mt] = wmma_f16(Af, Bf, acc[mt]);
            }
        }
        #pragma unroll
        for (int mt = 0; mt < 4; ++mt) {
            const int m0 = mt * 16 + hi * 8;
            #pragma unroll
            for (int j = 0; j < 8; ++j) {
                const int m = m0 + j;
                if (m < NTOK)
                    aout[((size_t)b * NTOK + m) * DIM + h * HD + d] =
                        (_Float16)(acc[mt][j] * R[m]);
            }
        }
    }
}

// ---------------- kernel 3: output projection, f32 out ----------------
__global__ __launch_bounds__(256) void proj_kernel(
    const _Float16* __restrict__ aout, const _Float16* __restrict__ wp16,
    const float* __restrict__ proj_b, float* __restrict__ out)
{
    const int wave = threadIdx.x >> 5, lane = threadIdx.x & 31;
    const int lh = lane & 15, hi = lane >> 4;
    const size_t mbase = ((size_t)blockIdx.x * 8 + wave) * 16;   // 12544 M-tiles exactly
    const int k0 = hi * 8;

    v16h Af[6];
    #pragma unroll
    for (int kt = 0; kt < 6; ++kt)
        Af[kt] = load_frag(aout + (mbase + lh) * DIM + kt * 32 + k0);

    #pragma unroll
    for (int nt = 0; nt < 12; ++nt) {
        const int o = nt * 16 + lh;
        v8f acc = {};
        #pragma unroll
        for (int kt = 0; kt < 6; ++kt) {
            v16h Bf = load_frag(wp16 + (size_t)o * DIM + kt * 32 + k0);
            acc = wmma_f16(Af[kt], Bf, acc);
        }
        const float bias = proj_b[o];
        #pragma unroll
        for (int j = 0; j < 8; ++j)
            out[(mbase + hi * 8 + j) * DIM + o] = acc[j] + bias;
    }
}

extern "C" void kernel_launch(void* const* d_in, const int* in_sizes, int n_in,
                              void* d_out, int out_size, void* d_ws, size_t ws_size,
                              hipStream_t stream) {
    (void)in_sizes; (void)n_in; (void)out_size; (void)ws_size;
    const float* x      = (const float*)d_in[0];
    const float* mask   = (const float*)d_in[1];
    const float* qkv_w  = (const float*)d_in[2];
    const float* qkv_b  = (const float*)d_in[3];
    const float* proj_w = (const float*)d_in[4];
    const float* proj_b = (const float*)d_in[5];
    const float* rpb    = (const float*)d_in[6];
    const int*   relidx = (const int*)d_in[7];
    float* out = (float*)d_out;

    char* ws = (char*)d_ws;
    _Float16* wq16 = (_Float16*)(ws + OFF_WQ);
    _Float16* wp16 = (_Float16*)(ws + OFF_WP);
    float*    comb = (float*)   (ws + OFF_COMB);
    _Float16* Qb   = (_Float16*)(ws + OFF_Q);
    _Float16* Kb   = (_Float16*)(ws + OFF_K);
    _Float16* VTb  = (_Float16*)(ws + OFF_VT);
    _Float16* aout = (_Float16*)(ws + OFF_AOUT);

    setup_kernel<<<(NW * NH * NPAD * NPAD + 255) / 256, 256, 0, stream>>>(
        qkv_w, proj_w, rpb, relidx, mask, wq16, wp16, comb);
    qkv_kernel<<<BWIN, 192, 0, stream>>>(x, wq16, qkv_b, Qb, Kb, VTb);
    attn_kernel<<<BWIN * NH / 2, 64, 0, stream>>>(Qb, Kb, VTb, comb, aout);
    proj_kernel<<<BWIN * NTOK / 16 / 8, 256, 0, stream>>>(aout, wp16, proj_b, out);
}
